// Neural_Network_2070174237162
// MI455X (gfx1250) — matrix operand order repaired and checked by reading, not compiled, not run on hardware
//
#include <hip/hip_runtime.h>
#include <hip/hip_bf16.h>

// Problem dims (match reference)
#define S_TOT  2048
#define A_TOT  512
#define IN_D   5
#define HID    256

// Tiling
#define SM     128            // structures per block tile (M)
#define LDH    264            // padded LDS row stride in bf16 elems (528 B, 16B-aligned)
#define STILES (S_TOT / SM)   // 16
#define SBLK   4              // grid.x; each block loops tiles with stride gridDim.x

typedef __attribute__((ext_vector_type(16))) __bf16 v16bf;
typedef __attribute__((ext_vector_type(8)))  float  v8f;

typedef __attribute__((address_space(1))) int global_i32;
typedef __attribute__((address_space(3))) int lds_i32;

union Frag {
    v16bf v;
    unsigned int u[8];
    unsigned short h[16];
};

// ---- Async global->LDS path (gfx1250), guarded so compile never breaks ----
#if __has_builtin(__builtin_amdgcn_global_load_async_to_lds_b32) && \
    __has_builtin(__builtin_amdgcn_s_wait_asynccnt)
#define HAVE_ASYNC_LDS 1
#endif

// ---- Fast scalar ops: gfx1250 hardware tanh / packed bf16 cvt when available ----
__device__ __forceinline__ float fast_tanh(float x) {
#if __has_builtin(__builtin_amdgcn_tanhf)
    return __builtin_amdgcn_tanhf(x);          // v_tanh_f32 (TRANS, co-executes with WMMA)
#else
    return tanhf(x);
#endif
}

__device__ __forceinline__ unsigned int pack2(float lo, float hi) {
#if __has_builtin(__builtin_amdgcn_cvt_pk_bf16_f32)
    auto p = __builtin_amdgcn_cvt_pk_bf16_f32(lo, hi);   // v_cvt_pk_bf16_f32
    return __builtin_bit_cast(unsigned int, p);
#else
    unsigned int ul = __float_as_uint(lo);
    ul += 0x7FFFu + ((ul >> 16) & 1u);                   // RNE
    unsigned int uh = __float_as_uint(hi);
    uh += 0x7FFFu + ((uh >> 16) & 1u);
    return (ul >> 16) | (uh & 0xFFFF0000u);
#endif
}

__device__ __forceinline__ unsigned short f2bf(float f) {
    return (unsigned short)(pack2(f, f) & 0xFFFFu);
}
__device__ __forceinline__ float bf2f(unsigned short h) {
    return __uint_as_float(((unsigned int)h) << 16);
}

// One hidden layer: [SM x 256](bf16, LDS) x [256 x 256](f32, global) -> tanh -> [SM x 256](bf16, LDS)
// 8 waves: wave w owns N-subtiles {2w, 2w+1} for all 8 M-subtiles.
__device__ __forceinline__
void layer_wmma(const unsigned short (*hin)[LDH],
                unsigned short (*hout)[LDH],
                const float* __restrict__ W,      // [HID][HID] row-major (k, n)
                const float* __restrict__ bias,   // [HID]
                int lane, int wave)
{
    const int nlane = lane & 15;          // B: column within 16-wide N tile
    const int kB    = (lane >> 4) * 16;   // B: K base for this half-wave
    const int mlane = lane & 15;          // A: row within 16-high M tile
    const int kA    = (lane >> 4) * 16;   // A: K base for this half-wave

    const v8f zero = {0.f, 0.f, 0.f, 0.f, 0.f, 0.f, 0.f, 0.f};
    v8f acc[8][2];
#pragma unroll
    for (int m = 0; m < 8; ++m) { acc[m][0] = zero; acc[m][1] = zero; }

    for (int k0 = 0; k0 < HID; k0 += 32) {
        // B fragments: gather W columns, convert f32 -> bf16 pairs.
        Frag bfr[2];
#pragma unroll
        for (int j = 0; j < 2; ++j) {
            const int n = (2 * wave + j) * 16 + nlane;
            const float* wp = W + (size_t)(k0 + kB) * HID + n;
#pragma unroll
            for (int i = 0; i < 8; ++i)
                bfr[j].u[i] = pack2(wp[(size_t)(2 * i) * HID],
                                    wp[(size_t)(2 * i + 1) * HID]);
        }

        // Batch-load all 8 A fragments first so the ds_load_b128s clause up and
        // the scheduler can hide LDS latency under WMMA issue (decrementing waits).
        Frag afr[8];
#pragma unroll
        for (int m = 0; m < 8; ++m) {
            const unsigned short* ap = &hin[m * 16 + mlane][k0 + kA];
            const uint4 q0 = *(const uint4*)ap;
            const uint4 q1 = *(const uint4*)(ap + 8);
            afr[m].u[0] = q0.x; afr[m].u[1] = q0.y; afr[m].u[2] = q0.z; afr[m].u[3] = q0.w;
            afr[m].u[4] = q1.x; afr[m].u[5] = q1.y; afr[m].u[6] = q1.z; afr[m].u[7] = q1.w;
        }
#pragma unroll
        for (int m = 0; m < 8; ++m) {
            acc[m][0] = __builtin_amdgcn_wmma_f32_16x16x32_bf16(
                false, afr[m].v, false, bfr[0].v, (short)0, acc[m][0], false, false);
            acc[m][1] = __builtin_amdgcn_wmma_f32_16x16x32_bf16(
                false, afr[m].v, false, bfr[1].v, (short)0, acc[m][1], false, false);
        }
    }

    // Epilogue: bias + tanh (v_tanh_f32), pack to bf16, scatter to LDS.
    // C/D layout: VGPR r, lane l -> row = (l>>4)*8 + r, col = l&15 (within 16x16 tile).
#pragma unroll
    for (int j = 0; j < 2; ++j) {
        const int n  = (2 * wave + j) * 16 + nlane;
        const float bv = bias[n];
#pragma unroll
        for (int m = 0; m < 8; ++m) {
            const int r0 = m * 16 + (lane >> 4) * 8;
#pragma unroll
            for (int r = 0; r < 8; ++r)
                hout[r0 + r][n] = f2bf(fast_tanh(acc[m][j][r] + bv));
        }
    }
}

__global__ void __launch_bounds__(256)
atom_mlp_kernel(const float* __restrict__ g,
                const float* __restrict__ W0, const float* __restrict__ b0,
                const float* __restrict__ W1, const float* __restrict__ b1,
                const float* __restrict__ W2, const float* __restrict__ b2,
                const float* __restrict__ W3, const float* __restrict__ b3,
                float* __restrict__ out)
{
    __shared__ __align__(16) unsigned short sh_a[SM][LDH];  // bf16 activations (ping)
    __shared__ __align__(16) unsigned short sh_b[SM][LDH];  // bf16 activations (pong)
    __shared__ __align__(16) float sg[SM][IN_D];            // staged g tile

    const int a    = blockIdx.y;
    const int tid  = threadIdx.x;
    const int lane = tid & 31;
    const int wave = tid >> 5;

    const float* W1a = W1 + (size_t)a * HID * HID;
    const float* W2a = W2 + (size_t)a * HID * HID;
    const float* b1a = b1 + (size_t)a * HID;
    const float* b2a = b2 + (size_t)a * HID;

    // Per-thread layer-0 weights/bias (column = tid), reused across all tiles.
    float w0c[IN_D];
#pragma unroll
    for (int i = 0; i < IN_D; ++i)
        w0c[i] = W0[((size_t)a * IN_D + i) * HID + tid];
    const float b0c = b0[(size_t)a * HID + tid];
    const float b3c = b3[a];
    const float* w3p = W3 + (size_t)a * HID;

    for (int tile = blockIdx.x; tile < STILES; tile += gridDim.x) {
        const int s0 = tile * SM;
        __syncthreads();  // protect LDS reuse from previous tile's readers

        // ---- Stage g tile into LDS ----
#ifdef HAVE_ASYNC_LDS
        for (int i = tid; i < SM * IN_D; i += 256) {
            const int row = i / IN_D, c = i % IN_D;
            const float* gp = &g[(size_t)(s0 + row) * A_TOT * IN_D + (size_t)a * IN_D + c];
            __builtin_amdgcn_global_load_async_to_lds_b32(
                (global_i32*)const_cast<float*>(gp),
                (lds_i32*)&sg[row][c],
                /*offset=*/0, /*cpol=*/0);
        }
        __builtin_amdgcn_s_wait_asynccnt(0);
#else
        for (int i = tid; i < SM * IN_D; i += 256) {
            const int row = i / IN_D, c = i % IN_D;
            sg[row][c] = g[(size_t)(s0 + row) * A_TOT * IN_D + (size_t)a * IN_D + c];
        }
#endif
        __syncthreads();

        // ---- Layer 0: [SM x 5] x [5 x 256] + b0, tanh -> sh_a (bf16) ----
        for (int row = 0; row < SM; ++row) {
            float acc = b0c;
#pragma unroll
            for (int i = 0; i < IN_D; ++i)
                acc = fmaf(sg[row][i], w0c[i], acc);
            sh_a[row][tid] = f2bf(fast_tanh(acc));
        }

        // Prefetch next layer's weights while layer 1 runs (global_prefetch_b8).
        {
            const char* p = (const char*)W2a;
            for (size_t off = (size_t)tid * 128; off < (size_t)HID * HID * 4; off += 256 * 128)
                __builtin_prefetch(p + off, 0, 1);
        }
        __syncthreads();

        // ---- Layer 1: sh_a -> sh_b ----
        layer_wmma(sh_a, sh_b, W1a, b1a, lane, wave);
        __syncthreads();

        // ---- Layer 2: sh_b -> sh_a ----
        layer_wmma(sh_b, sh_a, W2a, b2a, lane, wave);
        __syncthreads();

        // ---- Layer 3: [SM x 256] . [256] + b3 -> out[s, a] ----
        if (tid < SM) {
            const int row = tid;
            float acc = b3c;
#pragma unroll 4
            for (int k = 0; k < HID; ++k)
                acc = fmaf(bf2f(sh_a[row][k]), w3p[k], acc);
            out[(size_t)(s0 + row) * A_TOT + a] = acc;
        }
    }
}

extern "C" void kernel_launch(void* const* d_in, const int* in_sizes, int n_in,
                              void* d_out, int out_size, void* d_ws, size_t ws_size,
                              hipStream_t stream) {
    (void)in_sizes; (void)n_in; (void)out_size; (void)d_ws; (void)ws_size;
    const float* g  = (const float*)d_in[0];
    const float* W0 = (const float*)d_in[1];
    const float* b0 = (const float*)d_in[2];
    const float* W1 = (const float*)d_in[3];
    const float* b1 = (const float*)d_in[4];
    const float* W2 = (const float*)d_in[5];
    const float* b2 = (const float*)d_in[6];
    const float* W3 = (const float*)d_in[7];
    const float* b3 = (const float*)d_in[8];
    float* out = (float*)d_out;

    dim3 grid(SBLK, A_TOT);
    dim3 block(256);
    hipLaunchKernelGGL(atom_mlp_kernel, grid, block, 0, stream,
                       g, W0, b0, W1, b1, W2, b2, W3, b3, out);
}
